// SwinTransformer_70901320123178
// MI455X (gfx1250) — compile-verified, matrix operand order checked by reading, not run on hardware
//
#include <hip/hip_runtime.h>
#include <math.h>

// ---------------------------------------------------------------------------
// Swin-T (attention-only blocks) forward for MI455X / gfx1250.
//
// Roofline: ~150 GF total vs 23.3 TB/s HBM -> memory bound. Activations kept
// bf16 end-to-end (halves HBM traffic); all weights converted once to
// *transposed* bf16 (~18 MB, stays in 192 MB L2) so every WMMA operand
// fragment is two contiguous 16-byte loads per lane (b128), no predication.
// All GEMM-shaped math on v_wmma_f32_16x16x32_bf16 (wave32, one 16x16 tile
// per wave, fp32 accumulation). Attention fuses scores+softmax+PV+residual.
// ---------------------------------------------------------------------------

typedef unsigned short u16;
typedef __attribute__((ext_vector_type(16))) __bf16 v16bf;
typedef __attribute__((ext_vector_type(8)))  float  v8f;

__device__ __forceinline__ float bf2f(u16 h) { return __uint_as_float(((unsigned)h) << 16); }
__device__ __forceinline__ u16 f2bf(float f) {
  unsigned u = __float_as_uint(f);
  u += 0x7fffu + ((u >> 16) & 1u);            // round-to-nearest-even
  return (u16)(u >> 16);
}

union FragU {
  uint4 q[2];
  v16bf v;
};

// Fragment = two contiguous 16-byte chunks (lane's kb pattern is
// hlf*8 + {0..7} and 16 + hlf*8 + {0..7} consecutive u16 elements).
__device__ __forceinline__ v16bf frag2(const u16* p0, const u16* p1) {
  FragU f;
  f.q[0] = *(const uint4*)p0;
  f.q[1] = *(const uint4*)p1;
  return f.v;
}

// ---------------------------------------------------------------------------
// fp32 (K,N) -> bf16 transposed (Npad,K), rows n>=N zero-filled.
// ---------------------------------------------------------------------------
__global__ void k_convert_t(const float* __restrict__ s, u16* __restrict__ d,
                            int K, int N, int Npad) {
  int t = blockIdx.x * 256 + threadIdx.x;
  if (t >= Npad * K) return;
  int n = t / K, k = t - n * K;
  d[t] = (n < N) ? f2bf(s[(size_t)k * N + n]) : (u16)0;
}

// ---------------------------------------------------------------------------
// Patch embed: 4x4/4 conv == per-patch GEMM with K=48.  (B*3136 threads)
// ---------------------------------------------------------------------------
__global__ void k_patch_embed(const float* __restrict__ x, const float* __restrict__ w,
                              const float* __restrict__ pb, u16* __restrict__ out) {
  int t = blockIdx.x * 256 + threadIdx.x;
  if (t >= 64 * 3136) return;
  int b = t / 3136, p = t - b * 3136;
  int py = p / 56, px = p - py * 56;
  float in[48];
  const float* xb = x + (size_t)b * 3 * 224 * 224;
#pragma unroll
  for (int ci = 0; ci < 3; ++ci)
#pragma unroll
    for (int ky = 0; ky < 4; ++ky)
#pragma unroll
      for (int kx = 0; kx < 4; ++kx)
        in[ci * 16 + ky * 4 + kx] =
            xb[((size_t)ci * 224 + (py * 4 + ky)) * 224 + (px * 4 + kx)];
  u16* op = out + (size_t)t * 96;
  for (int c = 0; c < 96; ++c) {
    float acc = pb[c];
    const float* wc = w + c * 48;
#pragma unroll
    for (int j = 0; j < 48; ++j) acc += in[j] * wc[j];
    op[c] = f2bf(acc);
  }
}

// ---------------------------------------------------------------------------
// bf16 WMMA GEMM:  Y[M,0..N) = gather(X)[M,K] @ Wt^T (+bias) * scale
// Wt is transposed bf16 (Npad x K), padded rows are zero.
// win=1: output row m in window order, input row gathered from spatial h.
// One wave per 16x16 tile, 4 waves / 128-thread block. No in-loop branches.
// ---------------------------------------------------------------------------
__global__ void k_gemm(const u16* __restrict__ X, const u16* __restrict__ Wt,
                       const float* __restrict__ bias, u16* __restrict__ Yb,
                       float* __restrict__ Yf, int M, int N, int K, float scale,
                       int win, int fs, int L) {
  int lane = threadIdx.x & 31;
  int wave = threadIdx.x >> 5;
  int mt = blockIdx.y * 4 + wave;
  int m0 = mt * 16;
  if (m0 >= M) return;                         // wave-uniform
  int n0 = blockIdx.x * 16;
  int hlf = lane >> 4;
  int li  = lane & 15;

  int m = m0 + li;                             // A row owned by this lane
  int row = m;
  if (win) {
    int b = m / L;
    int r = m - b * L;
    int wi = r / 49, t = r - wi * 49;
    int wpr = fs / 7;
    int wy = wi / wpr, wx = wi - wy * wpr;
    int ty = t / 7, tx = t - ty * 7;
    row = b * L + (wy * 7 + ty) * fs + wx * 7 + tx;
  }
  const u16* xp = X + (size_t)row * K + hlf * 8;
  int nc = n0 + li;
  const u16* wp = Wt + (size_t)nc * K + hlf * 8;   // transposed: contiguous K

  v8f acc;
#pragma unroll
  for (int r = 0; r < 8; ++r) acc[r] = 0.f;

  for (int k0 = 0; k0 < K; k0 += 32) {
    if (k0 + 64 <= K) {
      __builtin_prefetch(xp + k0 + 32, 0, 1);      // global_prefetch_b8
      __builtin_prefetch(wp + k0 + 32, 0, 1);
    }
    v16bf a = frag2(xp + k0, xp + k0 + 16);
    v16bf b = frag2(wp + k0, wp + k0 + 16);
    acc = __builtin_amdgcn_wmma_f32_16x16x32_bf16(false, a, false, b,
                                                  (short)0, acc, false, false);
  }
  if (nc >= N) return;                         // head padding columns
  float bv = bias ? bias[nc] : 0.f;
#pragma unroll
  for (int r = 0; r < 8; ++r) {
    int mr = m0 + hlf * 8 + r;                 // C/D layout: half selects M+8
    float val = (acc[r] + bv) * scale;
    size_t gi = (size_t)mr * N + nc;
    if (Yb) Yb[gi] = f2bf(val);
    else    Yf[gi] = val;
  }
}

// ---------------------------------------------------------------------------
// Fused window attention for one (batch, window, head):
//   LDS-stage q,k (row-major) and v transposed (lvT[d][key]) zero-padded,
//   scores = q k^T (WMMA) + rel-pos bias, masked softmax (half-wave shfl),
//   out = P v (WMMA), residual-add in place into spatial h (bf16 RMW).
// 128 threads = 4 waves; wave w owns query rows [16w,16w+16).
// All LDS fragment loads are two contiguous 16B chunks -> ds_load_b128.
// ---------------------------------------------------------------------------
__global__ void k_attn(const u16* __restrict__ Q, const u16* __restrict__ Km,
                       const u16* __restrict__ V, const float* __restrict__ btab,
                       u16* __restrict__ H, int nw, int C, int dph, int dpad,
                       int fs, int L) {
  extern __shared__ __align__(16) u16 sm[];
  u16* lq  = sm;                               // 64 x dpad (row=query)
  u16* lk  = lq + 64 * dpad;                   // 64 x dpad (row=key)
  u16* lvT = lk + 64 * dpad;                   // dpad x 64 (row=d, col=key)
  u16* prb = lvT + 64 * dpad;                  // 4 waves * 16 * 64

  int wg = blockIdx.x;
  int hd = wg & 7;
  int wi = (wg >> 3) % nw;
  int b  = wg / (nw * 8);
  int tid = threadIdx.x;

  size_t baseRow = (size_t)(b * nw + wi) * 49;
  int tot = 64 * dpad;                          // multiple of 128
  for (int idx = tid; idx < tot; idx += 128) {
    int r = idx / dpad;
    int d = idx - r * dpad;
    u16 q = 0, k = 0, v = 0;
    if (r < 49 && d < dph) {
      size_t g = (baseRow + r) * C + hd * dph + d;
      q = Q[g]; k = Km[g]; v = V[g];
    }
    lq[idx] = q;
    lk[idx] = k;
    lvT[d * 64 + r] = v;                       // transpose V while staging
  }
  __syncthreads();

  int wave = tid >> 5;
  int lane = tid & 31;
  int hlf  = lane >> 4;
  int li   = lane & 15;
  int m0   = wave * 16;

  v8f acc[4];
#pragma unroll
  for (int nt = 0; nt < 4; ++nt)
#pragma unroll
    for (int r = 0; r < 8; ++r) acc[nt][r] = 0.f;

  // scores: A = q (16 x dpad), B = k^T (dpad x 64)
  const u16* qp = lq + (m0 + li) * dpad + hlf * 8;
  for (int k0 = 0; k0 < dpad; k0 += 32) {
    v16bf a = frag2(qp + k0, qp + k0 + 16);
#pragma unroll
    for (int nt = 0; nt < 4; ++nt) {
      const u16* kp = lk + (nt * 16 + li) * dpad + hlf * 8;
      v16bf bb = frag2(kp + k0, kp + k0 + 16);
      acc[nt] = __builtin_amdgcn_wmma_f32_16x16x32_bf16(false, a, false, bb,
                                                        (short)0, acc[nt], false, false);
    }
  }

  // relative-position bias (recomputed: (dy+6)*13+(dx+6)) + key mask
#pragma unroll
  for (int nt = 0; nt < 4; ++nt) {
    int kc = nt * 16 + li;
#pragma unroll
    for (int r = 0; r < 8; ++r) {
      int qr = m0 + hlf * 8 + r;
      float s = acc[nt][r];
      if (kc >= 49) s = -1e30f;
      else if (qr < 49) {
        int dy = (qr / 7 - kc / 7) + 6;
        int dx = (qr % 7 - kc % 7) + 6;
        s += btab[(dy * 13 + dx) * 8 + hd];
      }
      acc[nt][r] = s;
    }
  }

  // softmax: row lives in 16 lanes (one half-wave) x 4 registers
#pragma unroll
  for (int r = 0; r < 8; ++r) {
    float mx = fmaxf(fmaxf(acc[0][r], acc[1][r]), fmaxf(acc[2][r], acc[3][r]));
#pragma unroll
    for (int d = 1; d < 16; d <<= 1) mx = fmaxf(mx, __shfl_xor(mx, d, 32));
    float sum = 0.f;
#pragma unroll
    for (int nt = 0; nt < 4; ++nt) {
      float e = __expf(acc[nt][r] - mx);
      acc[nt][r] = e;
      sum += e;
    }
#pragma unroll
    for (int d = 1; d < 16; d <<= 1) sum += __shfl_xor(sum, d, 32);
    float inv = 1.f / sum;
#pragma unroll
    for (int nt = 0; nt < 4; ++nt) acc[nt][r] *= inv;
  }

  // spill probabilities (C-layout) to LDS so we can reload them in A-layout
  u16* mp = prb + wave * 16 * 64;
#pragma unroll
  for (int nt = 0; nt < 4; ++nt) {
    int kc = nt * 16 + li;
#pragma unroll
    for (int r = 0; r < 8; ++r)
      mp[(hlf * 8 + r) * 64 + kc] = f2bf(acc[nt][r]);
  }
  __syncthreads();

  // out = P (16x64) @ v (64xdpad); residual add in place into spatial h
  int wpr = fs / 7;
  int wy = wi / wpr, wx = wi - wy * wpr;
  int ndt = dpad / 16;
  const u16* ap = mp + li * 64 + hlf * 8;
  for (int dt = 0; dt < ndt; ++dt) {
    v8f o;
#pragma unroll
    for (int r = 0; r < 8; ++r) o[r] = 0.f;
    const u16* vp = lvT + (dt * 16 + li) * 64 + hlf * 8;
#pragma unroll
    for (int kt = 0; kt < 2; ++kt) {
      v16bf a  = frag2(ap + kt * 32, ap + kt * 32 + 16);
      v16bf bb = frag2(vp + kt * 32, vp + kt * 32 + 16);
      o = __builtin_amdgcn_wmma_f32_16x16x32_bf16(false, a, false, bb,
                                                  (short)0, o, false, false);
    }
    int d = dt * 16 + li;
    if (d < dph) {
      int gc = hd * dph + d;
#pragma unroll
      for (int r = 0; r < 8; ++r) {
        int qr = m0 + hlf * 8 + r;
        if (qr < 49) {
          int ty = qr / 7, tx = qr - ty * 7;
          size_t sr = (size_t)b * L + (size_t)(wy * 7 + ty) * fs + wx * 7 + tx;
          size_t gi = sr * C + gc;
          H[gi] = f2bf(bf2f(H[gi]) + o[r]);     // exclusive (row,col) owner
        }
      }
    }
  }
}

// ---------------------------------------------------------------------------
// Patch-merge gather + LayerNorm. One wave per output row.
// Output row index = b*L2 + (x2*s2 + y2)  (folds the reference's H/W swap).
// ---------------------------------------------------------------------------
__global__ void k_merge_ln(const u16* __restrict__ Hin, const float* __restrict__ lnw,
                           const float* __restrict__ lnb, u16* __restrict__ out,
                           int s, int C, int s2) {
  int gwave = (blockIdx.x * 128 + threadIdx.x) >> 5;
  int lane = threadIdx.x & 31;
  int L2 = s2 * s2;
  int b = gwave / L2;
  int rp = gwave - b * L2;
  int y2 = rp / s2;
  int x2 = rp - y2 * s2;
  int C4 = 4 * C;
  size_t inb = (size_t)b * s * s;

  float sum = 0.f, sq = 0.f;
  for (int c4 = lane; c4 < C4; c4 += 32) {
    int g = c4 / C, c = c4 - g * C;
    int dy = g & 1, dx = g >> 1;               // concat order (0,0),(1,0),(0,1),(1,1)
    size_t ir = inb + (size_t)(2 * y2 + dy) * s + (2 * x2 + dx);
    float v = bf2f(Hin[ir * C + c]);
    sum += v; sq += v * v;
  }
#pragma unroll
  for (int d = 1; d < 32; d <<= 1) {
    sum += __shfl_xor(sum, d, 32);
    sq  += __shfl_xor(sq,  d, 32);
  }
  float mean = sum / (float)C4;
  float var  = sq / (float)C4 - mean * mean;
  float rstd = rsqrtf(var + 1e-5f);
  size_t orow = (size_t)b * L2 + (size_t)(x2 * s2 + y2);
  for (int c4 = lane; c4 < C4; c4 += 32) {
    int g = c4 / C, c = c4 - g * C;
    int dy = g & 1, dx = g >> 1;
    size_t ir = inb + (size_t)(2 * y2 + dy) * s + (2 * x2 + dx);
    float v = bf2f(Hin[ir * C + c]);
    float o = (v - mean) * rstd * lnw[c4] + lnb[c4];
    out[orow * (size_t)C4 + c4] = f2bf(o);
  }
}

// ---------------------------------------------------------------------------
// Pooling with the reference's raw (B,49,768)->(B,768,7,7) reinterpretation:
// pooled[b,c] = mean of 49 consecutive elements at flat offset b*37632 + c*49.
// ---------------------------------------------------------------------------
__global__ void k_pool(const u16* __restrict__ Hin, u16* __restrict__ out) {
  int t = blockIdx.x * 256 + threadIdx.x;
  if (t >= 64 * 768) return;
  int b = t / 768, c = t - b * 768;
  const u16* p = Hin + (size_t)b * 37632 + (size_t)c * 49;
  float s = 0.f;
#pragma unroll
  for (int i = 0; i < 49; ++i) s += bf2f(p[i]);
  out[t] = f2bf(s * (1.f / 49.f));
}

// ---------------------------------------------------------------------------
// Host orchestration
// ---------------------------------------------------------------------------
static const int DEPH[4] = {2, 2, 6, 2};
static const int EMBH[4] = {96, 192, 384, 768};
static const int FSH[4]  = {56, 28, 14, 7};

struct SwinPtrs {
  const float* x; const float* patch_w; const float* patch_b;
  const float *qw[12], *qb[12], *kw[12], *kb[12], *vw[12], *vb[12], *bias[12];
  const float *ln_w[3], *ln_b[3], *red_w[3], *red_b[3];
  const float *head_w, *head_b;
};

static void parse_sorted(void* const* din, int i, SwinPtrs& p) {
  // jax tree_flatten order (dict keys sorted alphabetically)
  p.head_b = (const float*)din[i++];
  p.head_w = (const float*)din[i++];
  p.patch_b = (const float*)din[i++];
  p.patch_w = (const float*)din[i++];
  int bi = 0;
  for (int s = 0; s < 4; ++s) {
    for (int d = 0; d < DEPH[s]; ++d) {       // 'blocks' < 'merge'
      p.bias[bi] = (const float*)din[i++];
      p.kb[bi]   = (const float*)din[i++];
      p.kw[bi]   = (const float*)din[i++];
      p.qb[bi]   = (const float*)din[i++];
      p.qw[bi]   = (const float*)din[i++];
      p.vb[bi]   = (const float*)din[i++];
      p.vw[bi]   = (const float*)din[i++];
      ++bi;
    }
    if (s > 0) {
      p.ln_b[s - 1]  = (const float*)din[i++];
      p.ln_w[s - 1]  = (const float*)din[i++];
      p.red_b[s - 1] = (const float*)din[i++];
      p.red_w[s - 1] = (const float*)din[i++];
    }
  }
}

static void parse_insert(void* const* din, int i, SwinPtrs& p) {
  // dict insertion order
  p.patch_w = (const float*)din[i++];
  p.patch_b = (const float*)din[i++];
  int bi = 0;
  for (int s = 0; s < 4; ++s) {
    if (s > 0) {                               // 'merge' inserted before 'blocks'
      p.ln_w[s - 1]  = (const float*)din[i++];
      p.ln_b[s - 1]  = (const float*)din[i++];
      p.red_w[s - 1] = (const float*)din[i++];
      p.red_b[s - 1] = (const float*)din[i++];
    }
    for (int d = 0; d < DEPH[s]; ++d) {
      p.qw[bi]   = (const float*)din[i++];
      p.qb[bi]   = (const float*)din[i++];
      p.kw[bi]   = (const float*)din[i++];
      p.kb[bi]   = (const float*)din[i++];
      p.vw[bi]   = (const float*)din[i++];
      p.vb[bi]   = (const float*)din[i++];
      p.bias[bi] = (const float*)din[i++];
      ++bi;
    }
  }
  p.head_w = (const float*)din[i++];
  p.head_b = (const float*)din[i++];
}

static void launch_gemm(hipStream_t st, const u16* X, const u16* Wt, const float* bias,
                        u16* Yb, float* Yf, int M, int N, int Npad, int K, float scale,
                        int win, int fs, int L) {
  dim3 g(Npad / 16, (M + 63) / 64);
  k_gemm<<<g, 128, 0, st>>>(X, Wt, bias, Yb, Yf, M, N, K, scale, win, fs, L);
}

extern "C" void kernel_launch(void* const* d_in, const int* in_sizes, int n_in,
                              void* d_out, int out_size, void* d_ws, size_t ws_size,
                              hipStream_t stream) {
  (void)out_size; (void)ws_size;

  SwinPtrs p{};
  if (in_sizes[0] == 1000) {                   // fully sorted pytree, x last
    parse_sorted(d_in, 0, p);
    p.x = (const float*)d_in[n_in - 1];
  } else {
    p.x = (const float*)d_in[0];
    if (n_in > 1 && in_sizes[1] == 1000) parse_sorted(d_in, 1, p);
    else                                 parse_insert(d_in, 1, p);
  }

  // ---- workspace bump allocator ----
  char* base = (char*)d_ws;
  size_t off = 0;
  auto alloc = [&](size_t bytes) -> void* {
    void* r = base + off;
    off = (off + bytes + 255) & ~(size_t)255;
    return r;
  };
  const size_t AMAX = 19267584;                // 64*3136*96 elements
  u16* hA = (u16*)alloc(AMAX * 2);             // activations (bf16)
  u16* T1 = (u16*)alloc(AMAX * 2);             // q / LN temp / pooled
  u16* T2 = (u16*)alloc(AMAX * 2);             // k
  u16* T3 = (u16*)alloc(AMAX * 2);             // v

  static const int CB[12] = {96, 96, 192, 192, 384, 384, 384, 384, 384, 384, 768, 768};
  u16 *WQ[12], *WK[12], *WV[12], *WR[3], *WH;
  for (int bi = 0; bi < 12; ++bi) {
    size_t n = (size_t)CB[bi] * CB[bi];
    WQ[bi] = (u16*)alloc(n * 2);
    WK[bi] = (u16*)alloc(n * 2);
    WV[bi] = (u16*)alloc(n * 2);
  }
  for (int m = 0; m < 3; ++m) {
    size_t n = (size_t)4 * EMBH[m] * 2 * EMBH[m];
    WR[m] = (u16*)alloc(n * 2);
  }
  const int NHEAD_PAD = 1008;                  // 1000 -> multiple of 16
  WH = (u16*)alloc((size_t)NHEAD_PAD * 768 * 2);

  // ---- transposed bf16 weight conversion (hot in the 192 MB L2) ----
  auto convt = [&](const float* s, u16* d, int K, int N, int Npad) {
    size_t n = (size_t)Npad * K;
    k_convert_t<<<dim3((unsigned)((n + 255) / 256)), 256, 0, stream>>>(s, d, K, N, Npad);
  };
  for (int bi = 0; bi < 12; ++bi) {
    int C = CB[bi];
    convt(p.qw[bi], WQ[bi], C, C, C);
    convt(p.kw[bi], WK[bi], C, C, C);
    convt(p.vw[bi], WV[bi], C, C, C);
  }
  for (int m = 0; m < 3; ++m)
    convt(p.red_w[m], WR[m], 4 * EMBH[m], 2 * EMBH[m], 2 * EMBH[m]);
  convt(p.head_w, WH, 768, 1000, NHEAD_PAD);

  // ---- forward ----
  k_patch_embed<<<dim3((64 * 3136 + 255) / 256), 256, 0, stream>>>(
      p.x, p.patch_w, p.patch_b, hA);

  int bi = 0;
  for (int s = 0; s < 4; ++s) {
    int C = EMBH[s], fs = FSH[s];
    int Lc = fs * fs;
    int nwv = (fs / 7) * (fs / 7);
    if (s > 0) {
      int Cp = EMBH[s - 1], sprev = FSH[s - 1], s2 = fs, L2 = s2 * s2;
      k_merge_ln<<<dim3((64 * L2) / 4), 128, 0, stream>>>(
          hA, p.ln_w[s - 1], p.ln_b[s - 1], T1, sprev, Cp, s2);
      launch_gemm(stream, T1, WR[s - 1], p.red_b[s - 1], hA, nullptr,
                  64 * L2, 2 * Cp, 2 * Cp, 4 * Cp, 1.f, 0, 0, 0);
    }
    int dph = C / 8;
    int dpad = (dph + 31) & ~31;               // 12/24->32, 48->64, 96->96
    float qs = 1.f / sqrtf((float)dph);        // fold 1/sqrt(dph) into q
    int M = 64 * Lc;
    for (int d = 0; d < DEPH[s]; ++d) {
      launch_gemm(stream, hA, WQ[bi], p.qb[bi], T1, nullptr, M, C, C, C, qs, 1, fs, Lc);
      launch_gemm(stream, hA, WK[bi], p.kb[bi], T2, nullptr, M, C, C, C, 1.f, 1, fs, Lc);
      launch_gemm(stream, hA, WV[bi], p.vb[bi], T3, nullptr, M, C, C, C, 1.f, 1, fs, Lc);
      size_t shb = (size_t)(3 * 64 * dpad + 4 * 16 * 64) * sizeof(u16);
      k_attn<<<dim3(64 * nwv * 8), 128, shb, stream>>>(
          T1, T2, T3, p.bias[bi], hA, nwv, C, dph, dpad, fs, Lc);
      ++bi;
    }
  }

  k_pool<<<dim3((64 * 768 + 255) / 256), 256, 0, stream>>>(hA, T1);
  launch_gemm(stream, T1, WH, p.head_b, nullptr, (float*)d_out,
              64, 1000, NHEAD_PAD, 768, 1.f, 0, 0, 0);
}
